// TransformerBlock_47442208751683
// MI455X (gfx1250) — matrix operand order repaired and checked by reading, not compiled, not run on hardware
//
#include <hip/hip_runtime.h>
#include <hip/hip_bf16.h>

// ---------------------------------------------------------------------------
// Types
// ---------------------------------------------------------------------------
typedef __attribute__((ext_vector_type(16))) __bf16 v16bf;
typedef __attribute__((ext_vector_type(8)))  __bf16 v8bf;
typedef __attribute__((ext_vector_type(4)))  __bf16 v4bf;
typedef __attribute__((ext_vector_type(8)))  float  v8f;
typedef __attribute__((ext_vector_type(4)))  float  v4f;
typedef __attribute__((ext_vector_type(4)))  unsigned int v4u;
typedef __attribute__((ext_vector_type(8)))  int v8i;
typedef __attribute__((ext_vector_type(4)))  int v4i;

__device__ __forceinline__ unsigned short f2bf(float x) {
    unsigned u = __float_as_uint(x);
    u += 0x7fffu + ((u >> 16) & 1u);   // round-to-nearest-even
    return (unsigned short)(u >> 16);
}

union V16 {                       // one WMMA 16-bit operand fragment per lane
    v16bf bf;
    v8bf  h8[2];
    v4bf  h4[4];
    unsigned short u[16];
    v4u q[2];                     // 2 x 16B
};

union V8P {                       // 8 bf16 <-> one 16B store
    v8bf h;
    v4u  q;
};

// ---------------------------------------------------------------------------
// Tensor Data Mover: DMA a [rows x cols] bf16 panel (row stride ldbElems) from
// global memory into LDS (rows packed contiguously, row stride = cols).
// D# per CDNA5 ISA ch.8: group0 = {count, lds_addr, global_addr, type=2},
// group1 = {data_size=2B, tensor_dim0/1, tile_dim0/1, tensor_dim0_stride}.
// This toolchain exposes the 6-arg builtin form:
//   (uint32x4 g0, int32x8 g1, int32x4 g2, int32x4 g3, int32x8 g4, i32 cpol)
// ---------------------------------------------------------------------------
#if defined(__AMDGCN__) && __has_builtin(__builtin_amdgcn_tensor_load_to_lds) && \
    __has_builtin(__builtin_amdgcn_s_wait_tensorcnt)
#define USE_TDM 1
#else
#define USE_TDM 0
#endif

#if USE_TDM
__device__ __forceinline__ void tdm_load_panel(const unsigned short* gaddr,
                                               unsigned ldsByteOff,
                                               int rows, int cols, int ldbElems)
{
    const unsigned long long ga = (unsigned long long)gaddr;
    v4u g0 = { 1u,                                   // count=1, user descriptor
               ldsByteOff,                           // LDS byte address
               (unsigned)(ga & 0xffffffffull),       // global_addr[31:0]
               (unsigned)((ga >> 32) & 0x01ffffffull) | 0x80000000u }; // [56:32] | type=2
    v8i g1 = { (int)(1u << 16),                                  // data_size=1 (2 bytes)
               (int)(((unsigned)cols & 0xffffu) << 16),          // tensor_dim0[15:0]
               (int)(((unsigned)rows & 0xffffu) << 16),          // dim0 hi=0 | tensor_dim1 lo
               (int)(((unsigned)cols & 0xffffu) << 16),          // dim1 hi=0 | tile_dim0
               (int)((unsigned)rows & 0xffffu),                  // tile_dim1 (tile_dim2=0)
               (int)ldbElems,                                    // tensor_dim0_stride lo32
               0, 0 };
    v4i g2 = { 0, 0, 0, 0 };
    v4i g3 = { 0, 0, 0, 0 };
    v8i g4 = { 0, 0, 0, 0, 0, 0, 0, 0 };
    __builtin_amdgcn_tensor_load_to_lds(g0, g1, g2, g3, g4, 0);
}
#endif

// ---------------------------------------------------------------------------
// Generic bf16 WMMA GEMM:  C[M x Nc] = act( scale * A[M x K] * Bt[Nc x K]^T + bias )
//   A  : bf16 row-major (global).  Bt : bf16 row-major "transposed-B".
//   The 64-row B panel is staged into LDS by the TDM (wave 0 issues
//   tensor_load_to_lds + s_wait_tensorcnt), in K-chunks of <=512 columns,
//   removing the 8x redundant per-wave global B loads.
//   Block = 256 threads = 8 waves; wave w -> rows [blockM+16w,+16), cols [blockN,+64)
//   grid.z batching: z0 = z/zdiv, z1 = z%zdiv with independent element strides.
//   All dims are exact multiples of the tile (128M x 64N x 32K) -> no bounds checks.
// ---------------------------------------------------------------------------
template<bool OUT_BF, bool RELU>
__global__ void __launch_bounds__(256)
gemm_bf16(const unsigned short* __restrict__ A_, long aS0, long aS1, int lda,
          const unsigned short* __restrict__ Bt_, long bS0, long bS1, int ldb,
          void* __restrict__ C_, long cS0, long cS1, int ldc,
          const float* __restrict__ bias, int K, float scale, int zdiv)
{
    extern __shared__ unsigned short ldsB[];     // 64 * min(K,512) bf16

    const int z  = blockIdx.z;
    const int z0 = z / zdiv, z1 = z % zdiv;
    const unsigned short* A  = A_ + (long)z0 * aS0 + (long)z1 * aS1;
    const long cOff = (long)z0 * cS0 + (long)z1 * cS1;

    const int w     = threadIdx.x >> 5;
    const int lane  = threadIdx.x & 31;
    const int mBase = blockIdx.y * 128 + w * 16;
    const int nBase = blockIdx.x * 64;
    const int rA    = lane & 15;          // A row within tile / B column within tile
    const int aHalf = lane >> 4;          // K-half select
    const int kbB   = aHalf * 8;

    // B panel start for this block: 64 rows from nBase
    const unsigned short* Bpanel =
        Bt_ + (long)z0 * bS0 + (long)z1 * bS1 + (size_t)nBase * ldb;

    v8f acc[4];
#pragma unroll
    for (int j = 0; j < 4; ++j)
#pragma unroll
        for (int t = 0; t < 8; ++t) acc[j][t] = 0.f;

    for (int kb = 0; kb < K; kb += 512) {
        const int kspan = (K - kb < 512) ? (K - kb) : 512;

        // ---- stage B panel chunk [64 x kspan] into LDS --------------------
#if USE_TDM
        if (threadIdx.x < 32) {           // wave 0 drives the DMA
            tdm_load_panel(Bpanel + kb, 0u, 64, kspan, ldb);
            __builtin_amdgcn_s_wait_tensorcnt(0);
        }
#else
        for (int i = threadIdx.x; i < 64 * kspan / 8; i += 256) {
            const int row = i / (kspan / 8), c8 = i % (kspan / 8);
            *(v4u*)(ldsB + (size_t)row * kspan + c8 * 8) =
                *(const v4u*)(Bpanel + kb + (size_t)row * ldb + c8 * 8);
        }
#endif
        __syncthreads();

        for (int k = 0; k < kspan; k += 32) {
            V16 a;
            const unsigned short* pa = A + (size_t)(mBase + rA) * lda + kb + k + aHalf * 8;
            a.q[0] = *(const v4u*)(pa);        // K = k + half*8 .. +7
            a.q[1] = *(const v4u*)(pa + 16);   // K = k + 16 + half*8 .. +7
#pragma unroll
            for (int j = 0; j < 4; ++j) {
                const unsigned short* pb = ldsB + (size_t)(j * 16 + rA) * kspan + k + kbB;
                V16 bv;
                bv.q[0] = *(const v4u*)(pb);
                bv.q[1] = *(const v4u*)(pb + 16);
                acc[j] = __builtin_amdgcn_wmma_f32_16x16x32_bf16(
                    false, a.bf, false, bv.bf, (short)0, acc[j], false, false);
            }
        }
        __syncthreads();                  // before the next chunk overwrites LDS
    }

#pragma unroll
    for (int j = 0; j < 4; ++j) {
        const int ncol = nBase + j * 16 + rA;
        const float bv = bias ? bias[ncol] : 0.f;
#pragma unroll
        for (int r = 0; r < 8; ++r) {
            const int m = mBase + r + 8 * aHalf;     // D layout: M = r + 8*(lane>>4)
            float val = acc[j][r] * scale + bv;
            if constexpr (RELU) val = fmaxf(val, 0.f);
            if constexpr (OUT_BF)
                ((unsigned short*)C_)[cOff + (size_t)m * ldc + ncol] = f2bf(val);
            else
                ((float*)C_)[cOff + (size_t)m * ldc + ncol] = val;
        }
    }
}

// ---------------------------------------------------------------------------
// Fused edge-logits + softmax:
//   logits[h,m] = qk[b,h,n,m] + sum_e qe[b,n,h,e]*edge[b,n,m,e]   (WMMA, heads
//   packed as the 16 columns, QK logits loaded as the C operand), then row
//   softmax over m held entirely in registers (16 x v8f per wave), cross-wave
//   max/sum via LDS, and bf16 attention written out.
//   One workgroup (4 waves) per (b,n); edge read exactly once for all heads.
// ---------------------------------------------------------------------------
__global__ void __launch_bounds__(128)
edge_attn_kernel(const float* __restrict__ edge,
                 const unsigned short* __restrict__ qe,   // [B*N, H*E] bf16
                 const float* __restrict__ qk,            // [B,H,N,N] f32 QK logits
                 unsigned short* __restrict__ attn)       // [B,H,N,N] bf16 out
{
    __shared__ float red[4][16];

    const int bn   = blockIdx.x;          // b*1024 + n
    const int b    = bn >> 10;
    const int n    = bn & 1023;
    const int w    = threadIdx.x >> 5;
    const int lane = threadIdx.x & 31;
    const int hcol = lane & 15;           // B column = head (A row within m-tile too)
    const int half = lane >> 4;
    const int kbB  = half * 8;

    // B operand (qe^T for this (b,n)), two K=32 steps covering e = 0..63
    V16 bq[2];
#pragma unroll
    for (int ks = 0; ks < 2; ++ks) {
        if (hcol < 8) {
            const unsigned short* p = qe + (size_t)bn * 512 + hcol * 64 + ks * 32 + kbB;
            bq[ks].q[0] = *(const v4u*)(p);
            bq[ks].q[1] = *(const v4u*)(p + 16);
        } else {
            bq[ks].q[0] = v4u{0u, 0u, 0u, 0u};
            bq[ks].q[1] = v4u{0u, 0u, 0u, 0u};
        }
    }

    const size_t rowBase = ((size_t)(b * 8 + (hcol & 7)) * 1024 + n) * 1024;
    const size_t edgeRow = (size_t)bn * 1024 * 64;

    v8f acc[16];                          // this wave's 16 m-tiles (m = 16*(w+4t))

#pragma unroll
    for (int t = 0; t < 16; ++t) {
        const int m0 = (w + 4 * t) * 16;

        if (t < 15) {                     // prefetch next tile's edge rows
            const int mN = (w + 4 * (t + 1)) * 16;
            const float* pn = edge + edgeRow + (size_t)(mN + hcol) * 64 + half * 8;
            __builtin_prefetch(pn, 0, 1);
            __builtin_prefetch(pn + 32, 0, 1);
        }

        // C operand initialized from the QK logits (adds the q.kn term for free)
        v8f c;
        if (hcol < 8) {
            const float* pq = qk + rowBase + m0 + 8 * half;
            v4f q0 = *(const v4f*)(pq);
            v4f q1 = *(const v4f*)(pq + 4);
            c[0] = q0.x; c[1] = q0.y; c[2] = q0.z; c[3] = q0.w;
            c[4] = q1.x; c[5] = q1.y; c[6] = q1.z; c[7] = q1.w;
        } else {
#pragma unroll
            for (int r = 0; r < 8; ++r) c[r] = 0.f;
        }

#pragma unroll
        for (int ks = 0; ks < 2; ++ks) {
            const float* pe = edge + edgeRow + (size_t)(m0 + hcol) * 64 + ks * 32 + half * 8;
            v4f f0 = *(const v4f*)(pe);
            v4f f1 = *(const v4f*)(pe + 4);
            v4f f2 = *(const v4f*)(pe + 16);
            v4f f3 = *(const v4f*)(pe + 20);
            V16 a;                                    // packed bf16 converts
            a.h4[0] = __builtin_convertvector(f0, v4bf);
            a.h4[1] = __builtin_convertvector(f1, v4bf);
            a.h4[2] = __builtin_convertvector(f2, v4bf);
            a.h4[3] = __builtin_convertvector(f3, v4bf);
            c = __builtin_amdgcn_wmma_f32_16x16x32_bf16(
                false, a.bf, false, bq[ks].bf, (short)0, c, false, false);
        }
        acc[t] = c;
    }

    // ---- softmax over m (row length 1024, spread across 4 waves) ----------
    float mx = -3.4e38f;
#pragma unroll
    for (int t = 0; t < 16; ++t)
#pragma unroll
        for (int r = 0; r < 8; ++r) mx = fmaxf(mx, acc[t][r]);
    mx = fmaxf(mx, __shfl_xor(mx, 16, 32));      // combine the two M-row halves
    if (lane < 16) red[w][lane] = mx;
    __syncthreads();
    mx = fmaxf(fmaxf(red[0][hcol], red[1][hcol]), fmaxf(red[2][hcol], red[3][hcol]));
    __syncthreads();

    float s = 0.f;
#pragma unroll
    for (int t = 0; t < 16; ++t)
#pragma unroll
        for (int r = 0; r < 8; ++r) {
            float e = __expf(acc[t][r] - mx);
            acc[t][r] = e;
            s += e;
        }
    s += __shfl_xor(s, 16, 32);
    if (lane < 16) red[w][lane] = s;
    __syncthreads();
    s = (red[0][hcol] + red[1][hcol]) + (red[2][hcol] + red[3][hcol]);
    const float inv = 1.f / s;

    // ---- normalize, convert to bf16, store attention ----------------------
    if (hcol < 8) {
#pragma unroll
        for (int t = 0; t < 16; ++t) {
            const int m0 = (w + 4 * t) * 16;
            v8f vv = acc[t] * inv;
            V8P p;
            p.h = __builtin_convertvector(vv, v8bf);
            *(v4u*)(attn + rowBase + m0 + 8 * half) = p.q;   // 16B store
        }
    }
}

// ---------------------------------------------------------------------------
// Residual + LayerNorm over dim=256. 8 waves/block, 1 wave per row.
// ---------------------------------------------------------------------------
template<bool WRITE_BF>
__global__ void __launch_bounds__(256)
resid_ln(const float* __restrict__ a, const float* __restrict__ b,
         const float* __restrict__ g, const float* __restrict__ beta,
         float* __restrict__ outF, unsigned short* __restrict__ outB)
{
    const int row  = blockIdx.x * 8 + (threadIdx.x >> 5);
    const int lane = threadIdx.x & 31;
    const float* pa = a + (size_t)row * 256;
    const float* pb = b + (size_t)row * 256;

    float v[8];
    float s = 0.f;
#pragma unroll
    for (int i = 0; i < 8; ++i) { int c = lane + 32 * i; v[i] = pa[c] + pb[c]; s += v[i]; }
#pragma unroll
    for (int o = 16; o > 0; o >>= 1) s += __shfl_xor(s, o, 32);
    const float mu = s * (1.f / 256.f);

    float vr = 0.f;
#pragma unroll
    for (int i = 0; i < 8; ++i) { float d = v[i] - mu; vr += d * d; }
#pragma unroll
    for (int o = 16; o > 0; o >>= 1) vr += __shfl_xor(vr, o, 32);
    const float rs = rsqrtf(vr * (1.f / 256.f) + 1e-6f);

#pragma unroll
    for (int i = 0; i < 8; ++i) {
        int c = lane + 32 * i;
        float y = (v[i] - mu) * rs * g[c] + beta[c];
        outF[(size_t)row * 256 + c] = y;
        if constexpr (WRITE_BF) outB[(size_t)row * 256 + c] = f2bf(y);
    }
}

// ---------------------------------------------------------------------------
// Packing / conversion helpers
// ---------------------------------------------------------------------------
__global__ void cvt_f32_bf16(const float* __restrict__ in, unsigned short* __restrict__ out, int n) {
    int i = blockIdx.x * blockDim.x + threadIdx.x;
    if (i < n) out[i] = f2bf(in[i]);
}

// out[b][c][r] = bf16(in[b*inBS + r*C + c])  (batched transpose + convert)
__global__ void transpose_cvt(const float* __restrict__ in, unsigned short* __restrict__ out,
                              int R, int C, long inBS, int n) {
    int i = blockIdx.x * blockDim.x + threadIdx.x;
    if (i >= n) return;
    int rc = R * C;
    int b = i / rc, rem = i % rc;
    int c = rem / R, r = rem % R;
    out[i] = f2bf(in[(long)b * inBS + (long)r * C + c]);
}

// Edge part of Wk: out[h][e][o] = bf16(Wk[h][(256+e)][o])
__global__ void cvt_wke(const float* __restrict__ Wk, unsigned short* __restrict__ out) {
    int i = blockIdx.x * blockDim.x + threadIdx.x;      // 8*64*64
    if (i >= 8 * 64 * 64) return;
    int h = i >> 12, rem = i & 4095, e = rem >> 6, o = rem & 63;
    out[i] = f2bf(Wk[(size_t)h * 20480 + (256 + e) * 64 + o]);
}

// vT[b][c][m] = v[b][m][c],  c = h*64+o in [0,512), m in [0,1024)
__global__ void transpose_v(const unsigned short* __restrict__ v, unsigned short* __restrict__ vT) {
    int i = blockIdx.x * blockDim.x + threadIdx.x;      // 2*512*1024
    if (i >= 2 * 512 * 1024) return;
    int b = i >> 19;
    int rem = i & ((1 << 19) - 1);
    int c = rem >> 10, m = rem & 1023;
    vT[i] = v[((size_t)b * 1024 + m) * 512 + c];
}

// ---------------------------------------------------------------------------
// Host launcher
// ---------------------------------------------------------------------------
static inline unsigned smemFor(int K) {
    int kspan = K < 512 ? K : 512;
    return (unsigned)(64 * kspan * 2);
}

extern "C" void kernel_launch(void* const* d_in, const int* in_sizes, int n_in,
                              void* d_out, int out_size, void* d_ws, size_t ws_size,
                              hipStream_t stream)
{
    (void)in_sizes; (void)n_in; (void)out_size; (void)ws_size;

    const float* node  = (const float*)d_in[0];
    const float* edge  = (const float*)d_in[1];
    const float* Wq    = (const float*)d_in[2];
    const float* Wk    = (const float*)d_in[3];
    const float* Wv    = (const float*)d_in[4];
    const float* Wp    = (const float*)d_in[5];
    const float* bp    = (const float*)d_in[6];
    const float* ln1_g = (const float*)d_in[7];
    const float* ln1_b = (const float*)d_in[8];
    const float* W1    = (const float*)d_in[9];
    const float* b1    = (const float*)d_in[10];
    const float* W2    = (const float*)d_in[11];
    const float* b2    = (const float*)d_in[12];
    const float* ln2_g = (const float*)d_in[13];
    const float* ln2_b = (const float*)d_in[14];

    char* ws = (char*)d_ws;
    size_t off = 0;
    auto alloc = [&](size_t bytes) -> char* {
        char* p = ws + off;
        off += (bytes + 255) & ~(size_t)255;
        return p;
    };

    unsigned short* node_bf = (unsigned short*)alloc((size_t)2048 * 256 * 2);
    unsigned short* WqT     = (unsigned short*)alloc((size_t)512 * 256 * 2);
    unsigned short* WkNT    = (unsigned short*)alloc((size_t)512 * 256 * 2);
    unsigned short* WvT     = (unsigned short*)alloc((size_t)512 * 256 * 2);
    unsigned short* WkE     = (unsigned short*)alloc((size_t)8 * 64 * 64 * 2);
    unsigned short* WpT     = (unsigned short*)alloc((size_t)256 * 512 * 2);
    unsigned short* W1T     = (unsigned short*)alloc((size_t)1024 * 256 * 2);
    unsigned short* W2T     = (unsigned short*)alloc((size_t)256 * 1024 * 2);
    unsigned short* q_bf    = (unsigned short*)alloc((size_t)2048 * 512 * 2);
    unsigned short* kn_bf   = (unsigned short*)alloc((size_t)2048 * 512 * 2);
    unsigned short* v_bf    = (unsigned short*)alloc((size_t)2048 * 512 * 2);
    unsigned short* qe_bf   = (unsigned short*)alloc((size_t)2048 * 512 * 2);
    unsigned short* vT_bf   = (unsigned short*)alloc((size_t)2 * 512 * 1024 * 2);
    float*          qk      = (float*)alloc((size_t)2 * 8 * 1024 * 1024 * 4);   // 64 MB
    unsigned short* attn_bf = (unsigned short*)alloc((size_t)2 * 8 * 1024 * 1024 * 2); // 32 MB
    unsigned short* mo_bf   = (unsigned short*)alloc((size_t)2048 * 512 * 2);
    float*          mha     = (float*)alloc((size_t)2048 * 256 * 4);
    float*          x_f     = (float*)alloc((size_t)2048 * 256 * 4);
    unsigned short* x_bf    = (unsigned short*)alloc((size_t)2048 * 256 * 2);
    unsigned short* ff_bf   = (unsigned short*)alloc((size_t)2048 * 1024 * 2);
    float*          ff2     = (float*)alloc((size_t)2048 * 256 * 4);

    // ---- pack weights / activations to bf16 -------------------------------
    cvt_f32_bf16<<<2048, 256, 0, stream>>>(node, node_bf, 2048 * 256);
    transpose_cvt<<<512, 256, 0, stream>>>(Wq, WqT, 256, 64, 16384, 8 * 256 * 64);
    transpose_cvt<<<512, 256, 0, stream>>>(Wk, WkNT, 256, 64, 20480, 8 * 256 * 64);
    transpose_cvt<<<512, 256, 0, stream>>>(Wv, WvT, 256, 64, 16384, 8 * 256 * 64);
    cvt_wke<<<128, 256, 0, stream>>>(Wk, WkE);
    transpose_cvt<<<512, 256, 0, stream>>>(Wp, WpT, 512, 256, 131072, 512 * 256);
    transpose_cvt<<<1024, 256, 0, stream>>>(W1, W1T, 256, 1024, 262144, 256 * 1024);
    transpose_cvt<<<1024, 256, 0, stream>>>(W2, W2T, 1024, 256, 262144, 1024 * 256);

    // ---- Q (pre-scaled 1/sqrt(HS)), Kn, V projections: [2048,256]x[256,512]
    gemm_bf16<true, false><<<dim3(8, 16, 1), 256, smemFor(256), stream>>>(
        node_bf, 0, 0, 256, WqT, 0, 0, 256, q_bf, 0, 0, 512, nullptr, 256, 0.125f, 1);
    gemm_bf16<true, false><<<dim3(8, 16, 1), 256, smemFor(256), stream>>>(
        node_bf, 0, 0, 256, WkNT, 0, 0, 256, kn_bf, 0, 0, 512, nullptr, 256, 1.f, 1);
    gemm_bf16<true, false><<<dim3(8, 16, 1), 256, smemFor(256), stream>>>(
        node_bf, 0, 0, 256, WvT, 0, 0, 256, v_bf, 0, 0, 512, nullptr, 256, 1.f, 1);

    // ---- qe[b,n,h,e] = q_h @ WkE_h^T  (per-head batch over grid.z) --------
    gemm_bf16<true, false><<<dim3(1, 16, 8), 256, smemFor(64), stream>>>(
        q_bf, 0, 64, 512, WkE, 0, 4096, 64, qe_bf, 0, 64, 512, nullptr, 64, 1.f, 8);

    // ---- QK logits: per (b,h)  [1024,64]x[64,1024] -> qk f32 --------------
    gemm_bf16<false, false><<<dim3(16, 8, 16), 256, smemFor(64), stream>>>(
        q_bf, 524288, 64, 512, kn_bf, 524288, 64, 512,
        qk, 8388608, 1048576, 1024, nullptr, 64, 1.f, 8);

    // ---- fused edge logits + softmax -> bf16 attention --------------------
    edge_attn_kernel<<<2048, 128, 0, stream>>>(edge, qe_bf, qk, attn_bf);

    // ---- v^T for the AV GEMM B-operand ------------------------------------
    transpose_v<<<4096, 256, 0, stream>>>(v_bf, vT_bf);

    // ---- attn @ V ---------------------------------------------------------
    gemm_bf16<true, false><<<dim3(1, 8, 16), 256, smemFor(1024), stream>>>(
        attn_bf, 8388608, 1048576, 1024, vT_bf, 524288, 65536, 1024,
        mo_bf, 524288, 64, 512, nullptr, 1024, 1.f, 8);

    // ---- output projection + bias ----------------------------------------
    gemm_bf16<false, false><<<dim3(4, 16, 1), 256, smemFor(512), stream>>>(
        mo_bf, 0, 0, 512, WpT, 0, 0, 512, mha, 0, 0, 256, bp, 512, 1.f, 1);

    // ---- residual + LN1 ---------------------------------------------------
    resid_ln<true><<<256, 256, 0, stream>>>(node, mha, ln1_g, ln1_b, x_f, x_bf);

    // ---- FFN --------------------------------------------------------------
    gemm_bf16<true, true><<<dim3(16, 16, 1), 256, smemFor(256), stream>>>(
        x_bf, 0, 0, 256, W1T, 0, 0, 256, ff_bf, 0, 0, 1024, b1, 256, 1.f, 1);
    gemm_bf16<false, false><<<dim3(4, 16, 1), 256, smemFor(1024), stream>>>(
        ff_bf, 0, 0, 1024, W2T, 0, 0, 1024, ff2, 0, 0, 256, b2, 1024, 1.f, 1);

    // ---- residual + LN2 -> output ----------------------------------------
    resid_ln<false><<<256, 256, 0, stream>>>(x_f, ff2, ln2_g, ln2_b, (float*)d_out, nullptr);
}
